// KronLinear_45921790329281
// MI455X (gfx1250) — matrix operand order repaired and checked by reading, not compiled, not run on hardware
//
#include <hip/hip_runtime.h>

// ---------------------------------------------------------------------------
// KronLinear on MI455X (gfx1250):
//   w = sum_r kron(s*a_r, b_r)  (1024 x 4096)
//   out = x @ w + bias          (16384 x 1024) @ (1024 x 4096)
// Split-precision bf16 WMMA GEMM (hi*hi + hi*lo + lo*hi ~= fp32) with
// double-buffered GLOBAL_LOAD_ASYNC_TO_LDS_B128 tile staging (ASYNCcnt).
// ---------------------------------------------------------------------------

typedef __attribute__((ext_vector_type(16))) __bf16 v16bf;
typedef __attribute__((ext_vector_type(8)))  __bf16 v8bf;
typedef __attribute__((ext_vector_type(4)))  __bf16 v4bf;
typedef __attribute__((ext_vector_type(8)))  float  v8f;

#define BATCH_M 16384
#define DIM_K   1024    // a1*b1 = 64*16
#define DIM_N   4096    // a2*b2 = 128*32
#define A1 64
#define A2 128
#define B1 16
#define B2 32
#define RANK 8

#define KSTEPS (DIM_K / 32)

// --- gfx1250 async copy: 16B global -> LDS per lane, tracked by ASYNCcnt ----
__device__ __forceinline__ void async_copy_b128(unsigned lds_addr, unsigned voff,
                                                const void* sbase) {
  asm volatile("global_load_async_to_lds_b128 %0, %1, %2"
               :: "v"(lds_addr), "v"(voff), "s"(sbase) : "memory");
}
#define WAIT_ASYNC(n) asm volatile("s_wait_asynccnt %0" :: "n"(n) : "memory")

// ---------------------------------------------------------------------------
// Kernel 1: build w, transposed [n][k], split into bf16 hi/lo.
//   w[(i*B1+kb)][(j*B2+l)] = sum_r (s[i][j]*a[r][i][j]) * b[r][kb][l]
// ---------------------------------------------------------------------------
__global__ __launch_bounds__(256) void build_w_kernel(
    const float* __restrict__ a, const float* __restrict__ b,
    const float* __restrict__ s,
    __bf16* __restrict__ wT_hi, __bf16* __restrict__ wT_lo) {
  int t = blockIdx.x * 256 + threadIdx.x;       // 0 .. 4M-1
  int r_idx = t & (DIM_K - 1);                  // k index (fastest -> coalesced)
  int c_idx = t >> 10;                          // n index
  int i  = r_idx >> 4;
  int kb = r_idx & 15;
  int j  = c_idx >> 5;
  int l  = c_idx & 31;

  float sv  = s[i * A2 + j];
  float acc = 0.0f;
#pragma unroll
  for (int r = 0; r < RANK; ++r) {
    float av = a[((size_t)(r * A1 + i)) * A2 + j];
    float bv = b[((size_t)(r * B1 + kb)) * B2 + l];
    acc += sv * av * bv;
  }
  __bf16 hi = (__bf16)acc;
  __bf16 lo = (__bf16)(acc - (float)hi);
  wT_hi[(size_t)c_idx * DIM_K + r_idx] = hi;
  wT_lo[(size_t)c_idx * DIM_K + r_idx] = lo;
}

// ---------------------------------------------------------------------------
// Kernel 2: split x (f32, [M][K]) into bf16 hi/lo, same layout.
// ---------------------------------------------------------------------------
__global__ __launch_bounds__(256) void split_x_kernel(
    const float* __restrict__ x,
    __bf16* __restrict__ x_hi, __bf16* __restrict__ x_lo) {
  size_t q = (size_t)blockIdx.x * 256 + threadIdx.x;   // float4 index
  float4 v = ((const float4*)x)[q];
  v4bf hi, lo;
  hi[0] = (__bf16)v.x; hi[1] = (__bf16)v.y; hi[2] = (__bf16)v.z; hi[3] = (__bf16)v.w;
  lo[0] = (__bf16)(v.x - (float)hi[0]);
  lo[1] = (__bf16)(v.y - (float)hi[1]);
  lo[2] = (__bf16)(v.z - (float)hi[2]);
  lo[3] = (__bf16)(v.w - (float)hi[3]);
  ((v4bf*)x_hi)[q] = hi;
  ((v4bf*)x_lo)[q] = lo;
}

// ---------------------------------------------------------------------------
// Kernel 3: GEMM. 128x128 block tile, BK=32, 256 threads = 8 wave32.
// LDS: 2 buffers x { A_hi, A_lo, B_hi, B_lo } x [128][32] bf16 = 64 KB.
// ---------------------------------------------------------------------------
__global__ __launch_bounds__(256) void kron_gemm_kernel(
    const __bf16* __restrict__ x_hi, const __bf16* __restrict__ x_lo,
    const __bf16* __restrict__ wT_hi, const __bf16* __restrict__ wT_lo,
    const float* __restrict__ bias, float* __restrict__ out) {

  __shared__ __align__(32) __bf16 smem[2 * 4 * 128 * 32];   // 64 KB
  const unsigned SEC  = 128 * 32 * 2;       // 8192 B per section
  const unsigned BUFB = 4 * SEC;            // 32768 B per buffer

  const int tid  = threadIdx.x;
  const int lane = tid & 31;
  const int wid  = tid >> 5;
  const int wm   = wid & 3;                 // 0..3 : 32-row strip
  const int wn   = wid >> 2;                // 0..1 : 64-col strip
  const int lh   = lane >> 4;
  const int ln   = lane & 15;

  const int blockM = blockIdx.y * 128;
  const int blockN = blockIdx.x * 128;

  // block-adjusted global bases (uniform -> SGPR pairs for GVS addressing)
  const __bf16* xh_b = x_hi  + (size_t)blockM * DIM_K;
  const __bf16* xl_b = x_lo  + (size_t)blockM * DIM_K;
  const __bf16* wh_b = wT_hi + (size_t)blockN * DIM_K;
  const __bf16* wl_b = wT_lo + (size_t)blockN * DIM_K;

  const unsigned lds0 = (unsigned)(size_t)&smem[0];

  // per-thread chunk geometry: 2 chunks of 16B per section per tile-step
  // chunk c = tid + 256*it : row = c>>2 (0..127), ko = (c&3)*8 halfs
  unsigned loff[2], goff[2];
#pragma unroll
  for (int it = 0; it < 2; ++it) {
    int c   = tid + 256 * it;
    int row = c >> 2;
    int ko  = (c & 3) * 8;
    loff[it] = (unsigned)(row * 64 + ko * 2);         // bytes in LDS tile
    goff[it] = (unsigned)(row * (DIM_K * 2) + ko * 2); // bytes in global row
  }

  v8f acc[2][4] = {};

  // ---- prologue: issue tile 0 into buffer 0 (8 async ops / thread) ----
#pragma unroll
  for (int it = 0; it < 2; ++it) {
    unsigned l0 = lds0 + loff[it];
    unsigned g0 = goff[it];
    async_copy_b128(l0 + 0 * SEC, g0, xh_b);
    async_copy_b128(l0 + 1 * SEC, g0, xl_b);
    async_copy_b128(l0 + 2 * SEC, g0, wh_b);
    async_copy_b128(l0 + 3 * SEC, g0, wl_b);
  }

  for (int ks = 0; ks < KSTEPS; ++ks) {
    // ---- issue next tile into the other buffer ----
    if (ks + 1 < KSTEPS) {
      unsigned buf = ((unsigned)(ks + 1) & 1) * BUFB;
      unsigned kb  = (unsigned)(ks + 1) * 64;          // 32 halfs = 64 bytes
#pragma unroll
      for (int it = 0; it < 2; ++it) {
        unsigned l0 = lds0 + buf + loff[it];
        unsigned g0 = goff[it] + kb;
        async_copy_b128(l0 + 0 * SEC, g0, xh_b);
        async_copy_b128(l0 + 1 * SEC, g0, xl_b);
        async_copy_b128(l0 + 2 * SEC, g0, wh_b);
        async_copy_b128(l0 + 3 * SEC, g0, wl_b);
      }
      WAIT_ASYNC(8);        // next tile's 8 ops may remain in flight
    } else {
      WAIT_ASYNC(0);
    }
    __syncthreads();        // current buffer visible to all waves

    // ---- compute from buffer ks&1 ----
    const __bf16* sA_hi = smem + (size_t)(ks & 1) * (4 * 128 * 32) + 0 * (128 * 32);
    const __bf16* sA_lo = sA_hi + 128 * 32;
    const __bf16* sB_hi = sA_hi + 2 * 128 * 32;
    const __bf16* sB_lo = sA_hi + 3 * 128 * 32;

    // B fragments: lane l -> col N=l%16, K = 8*(l/16) + 16*(h>>3) + (h&7)
    v16bf bh[4], bl[4];
#pragma unroll
    for (int nf = 0; nf < 4; ++nf) {
      int n_local = wn * 64 + nf * 16 + ln;
      int base = n_local * 32 + 8 * lh;
      v8bf b0h = *(const v8bf*)&sB_hi[base];
      v8bf b1h = *(const v8bf*)&sB_hi[base + 16];
      v8bf b0l = *(const v8bf*)&sB_lo[base];
      v8bf b1l = *(const v8bf*)&sB_lo[base + 16];
      bh[nf] = __builtin_shufflevector(b0h, b1h, 0, 1, 2, 3, 4, 5, 6, 7,
                                       8, 9, 10, 11, 12, 13, 14, 15);
      bl[nf] = __builtin_shufflevector(b0l, b1l, 0, 1, 2, 3, 4, 5, 6, 7,
                                       8, 9, 10, 11, 12, 13, 14, 15);
    }

    // A fragments: lane l -> row M=l%16, K = 8*(l/16) + 16*(h>>3) + (h&7)
#pragma unroll
    for (int mf = 0; mf < 2; ++mf) {
      int m_local = wm * 32 + mf * 16 + ln;
      int base0 = m_local * 32 + 8 * lh;
      v8bf a0h = *(const v8bf*)&sA_hi[base0];
      v8bf a1h = *(const v8bf*)&sA_hi[base0 + 16];
      v8bf a0l = *(const v8bf*)&sA_lo[base0];
      v8bf a1l = *(const v8bf*)&sA_lo[base0 + 16];
      v16bf ah = __builtin_shufflevector(a0h, a1h, 0, 1, 2, 3, 4, 5, 6, 7,
                                         8, 9, 10, 11, 12, 13, 14, 15);
      v16bf al = __builtin_shufflevector(a0l, a1l, 0, 1, 2, 3, 4, 5, 6, 7,
                                         8, 9, 10, 11, 12, 13, 14, 15);
#pragma unroll
      for (int nf = 0; nf < 4; ++nf) {
        acc[mf][nf] = __builtin_amdgcn_wmma_f32_16x16x32_bf16(
            false, ah, false, bh[nf], (short)0, acc[mf][nf], false, false);
        acc[mf][nf] = __builtin_amdgcn_wmma_f32_16x16x32_bf16(
            false, ah, false, bl[nf], (short)0, acc[mf][nf], false, false);
        acc[mf][nf] = __builtin_amdgcn_wmma_f32_16x16x32_bf16(
            false, al, false, bh[nf], (short)0, acc[mf][nf], false, false);
      }
    }
    __syncthreads();        // all waves done before buffer is overwritten
  }

  // ---- epilogue: C/D layout: vgpr r -> M = r + 8*(lane/16), N = lane%16
#pragma unroll
  for (int mf = 0; mf < 2; ++mf) {
#pragma unroll
    for (int nf = 0; nf < 4; ++nf) {
      int col = blockN + wn * 64 + nf * 16 + ln;
      float bv = bias[col];
      int row0 = blockM + wm * 32 + mf * 16 + 8 * lh;
#pragma unroll
      for (int r = 0; r < 8; ++r) {
        out[(size_t)(row0 + r) * DIM_N + col] = acc[mf][nf][r] + bv;
      }
    }
  }
}

// ---------------------------------------------------------------------------
extern "C" void kernel_launch(void* const* d_in, const int* in_sizes, int n_in,
                              void* d_out, int out_size, void* d_ws, size_t ws_size,
                              hipStream_t stream) {
  const float* x    = (const float*)d_in[0];
  const float* a    = (const float*)d_in[1];
  const float* b    = (const float*)d_in[2];
  const float* s    = (const float*)d_in[3];
  const float* bias = (const float*)d_in[4];
  float* out = (float*)d_out;

  // workspace layout (bf16): wT_hi | wT_lo | x_hi | x_lo  (~80 MB total)
  __bf16* wT_hi = (__bf16*)d_ws;
  __bf16* wT_lo = wT_hi + (size_t)DIM_N * DIM_K;
  __bf16* x_hi  = wT_lo + (size_t)DIM_N * DIM_K;
  __bf16* x_lo  = x_hi  + (size_t)BATCH_M * DIM_K;

  build_w_kernel<<<(DIM_N * DIM_K) / 256, 256, 0, stream>>>(a, b, s, wT_hi, wT_lo);
  split_x_kernel<<<(BATCH_M * DIM_K / 4) / 256, 256, 0, stream>>>(x, x_hi, x_lo);

  dim3 grid(DIM_N / 128, BATCH_M / 128);
  kron_gemm_kernel<<<grid, 256, 0, stream>>>(x_hi, x_lo, wT_hi, wT_lo, bias, out);
}
